// SelfAttention_71717363909073
// MI455X (gfx1250) — compile-verified, matrix operand order checked by reading, not run on hardware
//
#include <hip/hip_runtime.h>
#include <hip/hip_bf16.h>
#include <math.h>

// Problem constants (from reference): B=16, H=128, W=128, C=256, C8=32
#define NB 16
#define NH 128
#define NW 128
#define NC 256
#define NC8 32

typedef __attribute__((ext_vector_type(16))) __bf16 v16bf;
typedef __attribute__((ext_vector_type(8)))  __bf16 v8bf;
typedef __attribute__((ext_vector_type(8)))  float  v8f;
typedef __attribute__((ext_vector_type(4)))  float  v4f;

// ---------------------------------------------------------------------------
// Weight prep: convert fp32 weight [256 x ncols] into bf16 fragments laid out
// as [ktile][ntile][lane(32)][elem(16)] matching the CDNA5 16-bit B-matrix
// VGPR layout (lane = N%16 + 16*halfK; elem j -> K = half*8 + j (j<8) or
// 16 + half*8 + (j-8)).  One contiguous 32B chunk per lane per fragment.
// ---------------------------------------------------------------------------
__global__ void prep_weights_kernel(const float* __restrict__ w,
                                    __bf16* __restrict__ dst, int ncols) {
    int idx = blockIdx.x * blockDim.x + threadIdx.x;
    int ntiles = ncols >> 4;
    int total = 8 * ntiles * 32 * 16;   // ktiles(8) * ntiles * lanes * elems
    if (idx >= total) return;
    int j    = idx & 15;
    int lane = (idx >> 4) & 31;
    int frag = idx >> 9;                // kt*ntiles + nt
    int nt = frag % ntiles;
    int kt = frag / ntiles;
    int half = lane >> 4;
    int N    = lane & 15;
    int K = kt * 32 + half * 8 + ((j < 8) ? j : (16 + (j - 8)));
    dst[idx] = (__bf16)w[K * ncols + (nt * 16 + N)];
}

// ---------------------------------------------------------------------------
// Fused attention kernel: one workgroup (8 waves) per (b,h) row.
// Each wave owns M-tile `wave` (rows wave*16 .. wave*16+15) in every phase,
// so A-fragments are hoisted to registers once and reused across all tiles.
// ---------------------------------------------------------------------------
union ShMemXS {
    __bf16 xs[NW][NC + 8];   // x row in bf16   (alive: phases 1-3)
    float  ss[NW][NW + 4];   // attention scores (alive: phases 4-6)
};

__device__ __forceinline__ v16bf load_afrag_bf16(const __bf16* rowptr, int k0, int half) {
    // 16-bit A/B fragment: two contiguous 8-element runs per lane.
    const v8bf lo = *(const v8bf*)(rowptr + k0 + half * 8);
    const v8bf hi = *(const v8bf*)(rowptr + k0 + 16 + half * 8);
    v16bf r;
#pragma unroll
    for (int i = 0; i < 8; ++i) { r[i] = lo[i]; r[i + 8] = hi[i]; }
    return r;
}

__global__ __launch_bounds__(256) void attn_row_kernel(
    const float* __restrict__ x,
    const float* __restrict__ bq, const float* __restrict__ bk,
    const float* __restrict__ bv,
    const __bf16* __restrict__ wq_f, const __bf16* __restrict__ wk_f,
    const __bf16* __restrict__ wv_f,
    float* __restrict__ out)
{
    __shared__ __align__(16) ShMemXS U;                 // 67,584 B
    __shared__ __align__(16) __bf16 qs[NW][NC8 + 8];    // 10,240 B
    __shared__ __align__(16) __bf16 ks[NW][NC8 + 8];    // 10,240 B
    __shared__ __align__(16) __bf16 vst[NC][NW + 8];    // 69,632 B  (V transposed)
    __shared__ float red[2][256];                       //  2,048 B

    const int tid   = threadIdx.x;
    const int wave  = tid >> 5;          // 0..7 == M-tile owned by this wave
    const int lane  = tid & 31;
    const int laneN = lane & 15;
    const int laneH = lane >> 4;

    const size_t rowbase = (size_t)blockIdx.x * (NW * NC);

    // ---- Phase 1: load x row, convert fp32 -> bf16 into LDS -------------
    {
        const v4f* src = (const v4f*)(x + rowbase);
        for (int i = tid; i < (NW * NC / 4); i += 256) {
            v4f d = src[i];
            int f = i * 4;
            int r = f >> 8;          // / NC
            int c = f & (NC - 1);
            U.xs[r][c]     = (__bf16)d[0];
            U.xs[r][c + 1] = (__bf16)d[1];
            U.xs[r][c + 2] = (__bf16)d[2];
            U.xs[r][c + 3] = (__bf16)d[3];
        }
    }
    __syncthreads();

    // ---- Hoist A-fragments of this wave's 16 x-rows (all 8 K-tiles) -----
    v16bf xa[8];
#pragma unroll
    for (int kt = 0; kt < 8; ++kt)
        xa[kt] = load_afrag_bf16(&U.xs[wave * 16 + laneN][0], kt * 32, laneH);

    // ---- Phase 2: Q and K projections (4 accumulators share A) ----------
    {
        v8f aq0 = {}, aq1 = {}, ak0 = {}, ak1 = {};
#pragma unroll
        for (int kt = 0; kt < 8; ++kt) {
            v16bf b;
            b = *(const v16bf*)(wq_f + (((kt * 2 + 0) * 32 + lane) << 4));
            aq0 = __builtin_amdgcn_wmma_f32_16x16x32_bf16(false, xa[kt], false, b,
                                                          (short)0, aq0, false, false);
            b = *(const v16bf*)(wq_f + (((kt * 2 + 1) * 32 + lane) << 4));
            aq1 = __builtin_amdgcn_wmma_f32_16x16x32_bf16(false, xa[kt], false, b,
                                                          (short)0, aq1, false, false);
            b = *(const v16bf*)(wk_f + (((kt * 2 + 0) * 32 + lane) << 4));
            ak0 = __builtin_amdgcn_wmma_f32_16x16x32_bf16(false, xa[kt], false, b,
                                                          (short)0, ak0, false, false);
            b = *(const v16bf*)(wk_f + (((kt * 2 + 1) * 32 + lane) << 4));
            ak1 = __builtin_amdgcn_wmma_f32_16x16x32_bf16(false, xa[kt], false, b,
                                                          (short)0, ak1, false, false);
        }
        const float bq0 = bq[laneN], bq1 = bq[16 + laneN];
        const float bk0 = bk[laneN], bk1 = bk[16 + laneN];
#pragma unroll
        for (int r = 0; r < 8; ++r) {
            const int M = wave * 16 + r + laneH * 8;
            qs[M][laneN]      = (__bf16)(aq0[r] + bq0);
            qs[M][16 + laneN] = (__bf16)(aq1[r] + bq1);
            ks[M][laneN]      = (__bf16)(ak0[r] + bk0);
            ks[M][16 + laneN] = (__bf16)(ak1[r] + bk1);
        }
    }

    // ---- Phase 3: V projection, stored transposed; A reused 16x ---------
#pragma unroll 4
    for (int nt = 0; nt < 16; ++nt) {
        v8f acc = {};
#pragma unroll
        for (int kt = 0; kt < 8; ++kt) {
            v16bf b = *(const v16bf*)(wv_f + (((kt * 16 + nt) * 32 + lane) << 4));
            acc = __builtin_amdgcn_wmma_f32_16x16x32_bf16(false, xa[kt], false, b,
                                                          (short)0, acc, false, false);
        }
        const float bb = bv[nt * 16 + laneN];
        __bf16* vrow = &vst[nt * 16 + laneN][0];   // fixed C row (transposed)
        const int col0 = wave * 16 + laneH * 8;
#pragma unroll
        for (int r = 0; r < 8; ++r)
            vrow[col0 + r] = (__bf16)(acc[r] + bb);
    }
    __syncthreads();   // xs dead from here; ss may be written

    // ---- Phase 4: scores = q @ k^T  (Q fragment hoisted, 8 tiles) -------
    {
        v16bf qa = load_afrag_bf16(&qs[wave * 16 + laneN][0], 0, laneH);
#pragma unroll
        for (int nt = 0; nt < 8; ++nt) {
            v16bf b = load_afrag_bf16(&ks[nt * 16 + laneN][0], 0, laneH);
            v8f acc = {};
            acc = __builtin_amdgcn_wmma_f32_16x16x32_bf16(false, qa, false, b,
                                                          (short)0, acc, false, false);
#pragma unroll
            for (int r = 0; r < 8; ++r)
                U.ss[wave * 16 + r + laneH * 8][nt * 16 + laneN] = acc[r];
        }
    }
    __syncthreads();

    // ---- Phase 5: row-wise softmax over 128 cols (2 threads per row) ----
    {
        const int row = tid >> 1;
        const int seg = tid & 1;
        float* sr = &U.ss[row][seg * 64];
        float mx = -INFINITY;
#pragma unroll 8
        for (int i = 0; i < 64; ++i) mx = fmaxf(mx, sr[i]);
        red[0][tid] = mx;
        __syncthreads();
        mx = fmaxf(red[0][row * 2], red[0][row * 2 + 1]);
        float sum = 0.0f;
#pragma unroll 8
        for (int i = 0; i < 64; ++i) { float e = __expf(sr[i] - mx); sr[i] = e; sum += e; }
        red[1][tid] = sum;
        __syncthreads();
        const float inv = 1.0f / (red[1][row * 2] + red[1][row * 2 + 1]);
#pragma unroll 8
        for (int i = 0; i < 64; ++i) sr[i] *= inv;
    }
    __syncthreads();

    // ---- Phase 6: out = P @ V  (P fragments converted once, 16 tiles) ---
    {
        v16bf pa[4];
        {
            const float* rowp = &U.ss[wave * 16 + laneN][0];
#pragma unroll
            for (int kt = 0; kt < 4; ++kt) {
                const int kb = kt * 32 + laneH * 8;
#pragma unroll
                for (int i = 0; i < 8; ++i) {
                    pa[kt][i]     = (__bf16)rowp[kb + i];
                    pa[kt][i + 8] = (__bf16)rowp[kb + 16 + i];
                }
            }
        }
#pragma unroll 4
        for (int nt = 0; nt < 16; ++nt) {
            v8f acc = {};
#pragma unroll
            for (int kt = 0; kt < 4; ++kt) {
                v16bf b = load_afrag_bf16(&vst[nt * 16 + laneN][0], kt * 32, laneH);
                acc = __builtin_amdgcn_wmma_f32_16x16x32_bf16(false, pa[kt], false, b,
                                                              (short)0, acc, false, false);
            }
            float* op = out + rowbase + (nt * 16 + laneN);
#pragma unroll
            for (int r = 0; r < 8; ++r)
                op[(size_t)(wave * 16 + r + laneH * 8) * NC] = acc[r];
        }
    }
}

// ---------------------------------------------------------------------------
extern "C" void kernel_launch(void* const* d_in, const int* in_sizes, int n_in,
                              void* d_out, int out_size, void* d_ws, size_t ws_size,
                              hipStream_t stream) {
    const float* x  = (const float*)d_in[0];
    const float* wq = (const float*)d_in[1];
    const float* bq = (const float*)d_in[2];
    const float* wk = (const float*)d_in[3];
    const float* bk = (const float*)d_in[4];
    const float* wv = (const float*)d_in[5];
    const float* bv = (const float*)d_in[6];
    float* out = (float*)d_out;

    // Workspace: bf16 fragment-swizzled weights
    __bf16* wq_f = (__bf16*)d_ws;            //  8*2*32*16  = 8192 elems (16 KB)
    __bf16* wk_f = wq_f + 8 * 2 * 32 * 16;   //  8192 elems (16 KB)
    __bf16* wv_f = wk_f + 8 * 2 * 32 * 16;   //  8*16*32*16 = 65536 elems (128 KB)

    prep_weights_kernel<<<(8192 + 255) / 256, 256, 0, stream>>>(wq, wq_f, NC8);
    prep_weights_kernel<<<(8192 + 255) / 256, 256, 0, stream>>>(wk, wk_f, NC8);
    prep_weights_kernel<<<(65536 + 255) / 256, 256, 0, stream>>>(wv, wv_f, NC);

    attn_row_kernel<<<NB * NH, 256, 0, stream>>>(x, bq, bk, bv,
                                                 wq_f, wk_f, wv_f, out);
}
